// DiffusionTransformerLayer_21440476741699
// MI455X (gfx1250) — matrix operand order repaired and checked by reading, not compiled, not run on hardware
//
#include <hip/hip_runtime.h>
#include <hip/hip_bf16.h>

// ---------------------------------------------------------------------------
// Types for CDNA5 WMMA (wave32): v_wmma_f32_16x16x32_bf16
// ---------------------------------------------------------------------------
typedef __attribute__((ext_vector_type(16))) __bf16 bf16x16;
typedef __attribute__((ext_vector_type(8)))  float  fx8;
typedef __attribute__((ext_vector_type(8)))  int    ix8;
typedef __attribute__((ext_vector_type(4)))  int    ix4;

#define DEV __device__ __forceinline__

DEV unsigned short f2bf(float f) {
  unsigned int u = __builtin_bit_cast(unsigned int, f);
  u += 0x7FFFu + ((u >> 16) & 1u);            // round-to-nearest-even
  return (unsigned short)(u >> 16);
}
DEV float bf2f(unsigned short h) {
  unsigned int u = ((unsigned int)h) << 16;
  return __builtin_bit_cast(float, u);
}
DEV float sigmoidf_(float x) { return 1.0f / (1.0f + __expf(-x)); }

DEV bf16x16 pack_frag(ix4 lo, ix4 hi) {
  ix8 r;
  r[0] = lo[0]; r[1] = lo[1]; r[2] = lo[2]; r[3] = lo[3];
  r[4] = hi[0]; r[5] = hi[1]; r[6] = hi[2]; r[7] = hi[3];
  return __builtin_bit_cast(bf16x16, r);
}

// B fragment. Bt = B^T stored [N][K] row-major bf16.
// Lane layout: col n = lane&15, K = k0 + 8*(lane>>4) + (e&7) + 16*(e>>3)
// -> two 16-byte runs per lane (two b128 loads).
DEV bf16x16 load_b_frag(const unsigned short* __restrict__ Bt, int n_row, int K,
                        int k0, int lane) {
  const unsigned short* p = Bt + (size_t)n_row * K + k0 + ((lane >> 4) << 3);
  ix4 lo = *(const ix4*)(p);
  ix4 hi = *(const ix4*)(p + 16);
  return pack_frag(lo, hi);
}

// A fragment from row-major [M][K] bf16 (global or LDS).
// Lane layout: row m = lane&15, K = k0 + 8*(lane>>4) + (e&7) + 16*(e>>3)
DEV bf16x16 load_a_frag(const unsigned short* __restrict__ A, int m_row, int K,
                        int k0, int lane) {
  const unsigned short* p = A + (size_t)m_row * K + k0 + ((lane >> 4) << 3);
  ix4 lo = *(const ix4*)(p);
  ix4 hi = *(const ix4*)(p + 16);
  return pack_frag(lo, hi);
}

DEV fx8 wmma_bf16(bf16x16 a, bf16x16 b, fx8 c) {
  return __builtin_amdgcn_wmma_f32_16x16x32_bf16(false, a, false, b, (short)0,
                                                 c, false, false);
}

// ---------------------------------------------------------------------------
// WMMA GEMM: C[M][N] = A[M][K](bf16) * B (given as Bt=[N][K] bf16) + bias
// Block = 256 threads = 8 waves -> 64 x 128 output tile.
// Each wave owns one 16-col strip and 4 row sub-tiles: one B fragment is
// reused by 4 WMMAs. All 4 A fragments are loaded into distinct registers
// before the WMMAs so the ds_load_b128s clause together and the 4 matrix ops
// issue back-to-back instead of each waiting on its own LDS read.
// A chunk (64x32) is staged in double-buffered LDS shared by all waves.
// ---------------------------------------------------------------------------
__global__ __launch_bounds__(256) void wmma_gemm_bf16(
    const unsigned short* __restrict__ A,
    const unsigned short* __restrict__ Bt,
    const float* __restrict__ bias,
    float* __restrict__ C, int M, int N, int K) {
  __shared__ unsigned short As[2][64 * 32];   // 2 x 4 KB
  const int tid = threadIdx.x;
  const int lane = tid & 31;
  const int wv = tid >> 5;
  const int m0 = blockIdx.y << 6;
  const int ncol = (blockIdx.x << 7) + (wv << 4) + (lane & 15);
  const int srow = tid >> 2, sgrp = tid & 3;  // staging: 1 b128 per thread
  const int nchunks = K >> 5;

  {  // stage chunk 0: 64 rows x 16 dwords = 256 b128 loads
    const ix4* src = (const ix4*)(A + (size_t)(m0 + srow) * K + (sgrp << 3));
    ((ix4*)As[0])[tid] = *src;
  }

  fx8 acc[4] = {};
  for (int c = 0; c < nchunks; ++c) {
    __syncthreads();
    if (c + 1 < nchunks) {
      const ix4* src =
          (const ix4*)(A + (size_t)(m0 + srow) * K + ((c + 1) << 5) + (sgrp << 3));
      ((ix4*)As[(c + 1) & 1])[tid] = *src;
      __builtin_prefetch(Bt + (size_t)ncol * K + ((c + 1) << 5), 0, 1);
    }
    bf16x16 bfg = load_b_frag(Bt, ncol, K, c << 5, lane);
    bf16x16 af[4];
#pragma unroll
    for (int ms = 0; ms < 4; ++ms)
      af[ms] = load_a_frag((const unsigned short*)As[c & 1],
                           (ms << 4) + (lane & 15), 32, 0, lane);
#pragma unroll
    for (int ms = 0; ms < 4; ++ms)
      acc[ms] = wmma_bf16(af[ms], bfg, acc[ms]);
  }
  const float bi = bias ? bias[ncol] : 0.0f;
#pragma unroll
  for (int ms = 0; ms < 4; ++ms) {
    const int mb = m0 + (ms << 4) + ((lane >> 4) << 3);
#pragma unroll
    for (int r = 0; r < 8; ++r)
      C[(size_t)(mb + r) * N + ncol] = acc[ms][r] + bi;
  }
}

// ---------------------------------------------------------------------------
// Attention: one block (4 waves) per (head, batch, 16-query tile).
// Phase 1: S = qK^T + z   (WMMA, K-dim 64 = padded head dim, scale folded in q)
// Phase 2: row softmax over 1024 keys (wave shuffles); P written once as bf16
// Phase 3: O = P V        (WMMA, K-dim 1024; P fragments via ds_load_b128,
//                          two chunks in flight)
// LDS: 64 KB f32 scores + 32 KB bf16 probs = 96 KB (< 320 KB WGP budget).
// ---------------------------------------------------------------------------
__global__ __launch_bounds__(128) void attn_wmma(
    const unsigned short* __restrict__ qb,   // [B*H][N][64]  (pre-scaled)
    const unsigned short* __restrict__ kb,   // [B*H][N][64]
    const unsigned short* __restrict__ vt,   // [B*H][48][N]  (v transposed)
    const float* __restrict__ z,             // [H][B][N][N]
    unsigned short* __restrict__ o,          // [B*N][H*48] bf16
    int Bn, int Hn, int Nn) {
  __shared__ float sc[16 * 1024];            // scores
  __shared__ unsigned short pb[16 * 1024];   // probabilities (bf16)
  const int tid = threadIdx.x, lane = tid & 31, wv = tid >> 5;
  const int qt = blockIdx.x;
  const int h = blockIdx.y / Bn, b = blockIdx.y % Bn;
  const int bh = b * Hn + h;
  const unsigned short* qbase = qb + (size_t)bh * Nn * 64;
  const unsigned short* kbase = kb + (size_t)bh * Nn * 64;

  const int qrow = (qt << 4) + (lane & 15);
  const bf16x16 qf0 = load_a_frag(qbase, qrow, 64, 0, lane);   // loop invariant
  const bf16x16 qf1 = load_a_frag(qbase, qrow, 64, 32, lane);
  for (int kt = wv; kt < (Nn >> 4); kt += 4) {
    fx8 acc = {};
    bf16x16 kf0 = load_b_frag(kbase, (kt << 4) + (lane & 15), 64, 0, lane);
    bf16x16 kf1 = load_b_frag(kbase, (kt << 4) + (lane & 15), 64, 32, lane);
    acc = wmma_bf16(qf0, kf0, acc);
    acc = wmma_bf16(qf1, kf1, acc);
    const int key = (kt << 4) + (lane & 15);
    const size_t zb = ((size_t)(h * Bn + b) * Nn + (qt << 4)) * Nn;
#pragma unroll
    for (int r = 0; r < 8; ++r) {
      int qr = r + ((lane >> 4) << 3);
      sc[(qr << 10) + key] = acc[r] + z[zb + (size_t)qr * Nn + key];
    }
  }
  __syncthreads();

  for (int rr = 0; rr < 4; ++rr) {           // 4 rows per wave
    const int rowi = (wv << 2) + rr;
    float* sr = sc + (rowi << 10);
    unsigned short* pr = pb + (rowi << 10);
    float mx = -3.0e38f;
    for (int j = lane; j < Nn; j += 32) mx = fmaxf(mx, sr[j]);
    for (int ofs = 16; ofs; ofs >>= 1) mx = fmaxf(mx, __shfl_xor(mx, ofs, 32));
    float sum = 0.0f;
    for (int j = lane; j < Nn; j += 32) {
      float e = __expf(sr[j] - mx);
      sr[j] = e;
      sum += e;
    }
    for (int ofs = 16; ofs; ofs >>= 1) sum += __shfl_xor(sum, ofs, 32);
    const float inv = 1.0f / sum;
    for (int j = lane; j < Nn; j += 32) pr[j] = f2bf(sr[j] * inv);
  }
  __syncthreads();

  if (wv < 3) {                              // 3 x 16 output cols = head dim 48
    fx8 acc = {};
    const unsigned short* vbase = vt + (size_t)bh * 48 * Nn;
    for (int c = 0; c < (Nn >> 5); c += 2) { // two chunks in flight
      bf16x16 af0 = load_a_frag(pb, lane & 15, 1024, c << 5, lane);
      bf16x16 bf0 = load_b_frag(vbase, (wv << 4) + (lane & 15), Nn, c << 5, lane);
      bf16x16 af1 = load_a_frag(pb, lane & 15, 1024, (c + 1) << 5, lane);
      bf16x16 bf1 = load_b_frag(vbase, (wv << 4) + (lane & 15), Nn, (c + 1) << 5, lane);
      acc = wmma_bf16(af0, bf0, acc);
      acc = wmma_bf16(af1, bf1, acc);
    }
    const int dcol = h * 48 + (wv << 4) + (lane & 15);
#pragma unroll
    for (int r = 0; r < 8; ++r) {
      int qr = (qt << 4) + r + ((lane >> 4) << 3);
      o[(size_t)(b * Nn + qr) * (Hn * 48) + dcol] = f2bf(acc[r]);
    }
  }
}

// ---------------------------------------------------------------------------
// Elementwise / rowwise helpers
// ---------------------------------------------------------------------------
__global__ void cvt_transpose_kernel(const float* __restrict__ W,     // [K][N]
                                     unsigned short* __restrict__ Wt, // [N][K]
                                     int K, int N) {
  int i = blockIdx.x * blockDim.x + threadIdx.x;
  if (i >= K * N) return;
  int n = i % N, k = i / N;
  Wt[(size_t)n * K + k] = f2bf(W[i]);
}

__global__ void cvt_bf16_kernel(const float* __restrict__ src,
                                unsigned short* __restrict__ dst, int n) {
  int i = blockIdx.x * blockDim.x + threadIdx.x;
  if (i < n) dst[i] = f2bf(src[i]);
}

// One wave per row layernorm; optional weight, optional f32/bf16 outputs.
__global__ __launch_bounds__(256) void ln_kernel(
    const float* __restrict__ x, const float* __restrict__ w,
    float* __restrict__ y32, unsigned short* __restrict__ y16, int rows, int d) {
  const int wid = threadIdx.x >> 5, lane = threadIdx.x & 31;
  const int row = blockIdx.x * 8 + wid;
  if (row >= rows) return;
  const float* xr = x + (size_t)row * d;
  float s = 0.0f, s2 = 0.0f;
  for (int j = lane; j < d; j += 32) { float v = xr[j]; s += v; s2 += v * v; }
  for (int ofs = 16; ofs; ofs >>= 1) {
    s += __shfl_xor(s, ofs, 32);
    s2 += __shfl_xor(s2, ofs, 32);
  }
  const float m = s / d;
  const float rstd = rsqrtf(s2 / d - m * m + 1e-5f);
  for (int j = lane; j < d; j += 32) {
    float v = (xr[j] - m) * rstd;
    if (w) v *= w[j];
    if (y32) y32[(size_t)row * d + j] = v;
    if (y16) y16[(size_t)row * d + j] = f2bf(v);
  }
}

__global__ void adaln_combine_kernel(const float* __restrict__ an,
                                     const float* __restrict__ ss,
                                     const float* __restrict__ sb,
                                     unsigned short* __restrict__ outb, int n) {
  int i = blockIdx.x * blockDim.x + threadIdx.x;
  if (i < n) outb[i] = f2bf(an[i] * sigmoidf_(ss[i]) + sb[i]);
}

__global__ void repack_qkv_kernel(const float* __restrict__ qf,
                                  const float* __restrict__ kf,
                                  const float* __restrict__ vf,
                                  unsigned short* __restrict__ qb,
                                  unsigned short* __restrict__ kb,
                                  unsigned short* __restrict__ vt,
                                  int Bn, int Hn, int Nn, float scale) {
  const int i = blockIdx.x * blockDim.x + threadIdx.x;
  const int total = Bn * Hn * Nn * 64;
  if (i >= total) return;
  const int d = i & 63;
  const int n = (i >> 6) % Nn;
  const int h = (i >> 6) / Nn % Hn;
  const int b = i / (64 * Nn * Hn);
  float qv = 0.0f, kv = 0.0f;
  if (d < 48) {
    const size_t src = ((size_t)(b * Nn + n) * Hn + h) * 48 + d;
    qv = qf[src] * scale;
    kv = kf[src];
    vt[((size_t)(b * Hn + h) * 48 + d) * Nn + n] = f2bf(vf[src]);
  }
  qb[i] = f2bf(qv);
  kb[i] = f2bf(kv);
}

__global__ void gate_mul_kernel(const unsigned short* __restrict__ o,
                                const float* __restrict__ gf,
                                unsigned short* __restrict__ xin, int n) {
  int i = blockIdx.x * blockDim.x + threadIdx.x;
  if (i < n) xin[i] = f2bf(bf2f(o[i]) * sigmoidf_(gf[i]));
}

__global__ void residual_gate_kernel(const float* __restrict__ base,
                                     const float* __restrict__ gate_lin,
                                     const float* __restrict__ x,
                                     float* __restrict__ out, int n) {
  int i = blockIdx.x * blockDim.x + threadIdx.x;
  if (i < n) out[i] = base[i] + sigmoidf_(gate_lin[i]) * x[i];
}

__global__ void ffn_combine_kernel(const float* __restrict__ sw,  // [rows][2*hd]
                                   const float* __restrict__ ab,  // [rows][hd]
                                   unsigned short* __restrict__ hid, int rows, int hd) {
  int i = blockIdx.x * blockDim.x + threadIdx.x;
  if (i >= rows * hd) return;
  int rr = i / hd, c = i % hd;
  float u = sw[(size_t)rr * (2 * hd) + c];
  float g = sw[(size_t)rr * (2 * hd) + hd + c];
  hid[i] = f2bf(g * sigmoidf_(g) * u * ab[i]);
}

// ---------------------------------------------------------------------------
// Host orchestration
// ---------------------------------------------------------------------------
extern "C" void kernel_launch(void* const* d_in, const int* in_sizes, int n_in,
                              void* d_out, int out_size, void* d_ws, size_t ws_size,
                              hipStream_t stream) {
  (void)in_sizes; (void)n_in; (void)out_size; (void)ws_size;
  const int Bc = 2, Nc = 1024, Dc = 768, Hc = 16, HIDc = 1536;
  const int Rc = Bc * Nc;  // 2048

  const float* a_in      = (const float*)d_in[0];
  const float* s_in      = (const float*)d_in[1];
  const float* z_in      = (const float*)d_in[2];
  const float* snw1      = (const float*)d_in[3];
  const float* ssw1      = (const float*)d_in[4];
  const float* ssb1      = (const float*)d_in[5];
  const float* sbw1      = (const float*)d_in[6];
  const float* q_w       = (const float*)d_in[7];
  const float* q_b       = (const float*)d_in[8];
  const float* k_w       = (const float*)d_in[9];
  const float* v_w       = (const float*)d_in[10];
  const float* g_w       = (const float*)d_in[11];
  const float* o_w       = (const float*)d_in[12];
  const float* outproj_w = (const float*)d_in[13];
  const float* outproj_b = (const float*)d_in[14];
  const float* snw2      = (const float*)d_in[15];
  const float* ssw2      = (const float*)d_in[16];
  const float* ssb2      = (const float*)d_in[17];
  const float* sbw2      = (const float*)d_in[18];
  const float* swish_w   = (const float*)d_in[19];
  const float* a2b_w     = (const float*)d_in[20];
  const float* b2a_w     = (const float*)d_in[21];
  const float* op_w      = (const float*)d_in[22];
  const float* op_b      = (const float*)d_in[23];
  float* out = (float*)d_out;

  // --- workspace bump allocator (lifetime-based reuse) ---
  char* wsb = (char*)d_ws;
  size_t off = 0;
  auto bump = [&](size_t bytes) -> char* {
    char* p = wsb + off;
    off = (off + bytes + 255) & ~(size_t)255;
    return p;
  };
  typedef unsigned short u16;
  const size_t DD = (size_t)Dc * Dc;
  u16* wt_ssw   = (u16*)bump(DD * 2);
  u16* wt_sbw   = (u16*)bump(DD * 2);
  u16* wt_q     = (u16*)bump(DD * 2);
  u16* wt_k     = (u16*)bump(DD * 2);
  u16* wt_v     = (u16*)bump(DD * 2);
  u16* wt_g     = (u16*)bump(DD * 2);
  u16* wt_o     = (u16*)bump(DD * 2);
  u16* wt_outp  = (u16*)bump(DD * 2);
  u16* wt_ssw2  = (u16*)bump(DD * 2);
  u16* wt_sbw2  = (u16*)bump(DD * 2);
  u16* wt_op    = (u16*)bump(DD * 2);
  u16* wt_swish = (u16*)bump((size_t)Dc * 2 * HIDc * 2);
  u16* wt_a2b   = (u16*)bump((size_t)Dc * HIDc * 2);
  u16* wt_b2a   = (u16*)bump((size_t)HIDc * Dc * 2);

  const size_t RD = (size_t)Rc * Dc;
  u16* s_bf   = (u16*)bump(RD * 2);
  u16* sn_bf  = (u16*)bump(RD * 2);          // sn1, later sn2
  u16* b_bf   = (u16*)bump(RD * 2);          // b, later a2
  u16* q_bf   = (u16*)bump((size_t)Bc * Hc * Nc * 64 * 2);
  u16* k_bf   = (u16*)bump((size_t)Bc * Hc * Nc * 64 * 2);
  u16* v_t    = (u16*)bump((size_t)Bc * Hc * 48 * Nc * 2);
  u16* o_bf   = (u16*)bump(RD * 2);
  u16* xin_bf = (u16*)bump(RD * 2);
  u16* hid_bf = (u16*)bump((size_t)Rc * HIDc * 2);

  float* an_f  = (float*)bump(RD * 4);       // LN(a), later LN(a_new)
  float* ss_f  = (float*)bump(RD * 4);       // ss1, later ss2
  float* sb_f  = (float*)bump(RD * 4);       // sb1, later sb2
  float* big4  = (float*)bump(RD * 4 * 4);   // qf|kf|vf|gf, later sw[R][3072]
  float* x_f   = (float*)bump(RD * 4);       // x, later hb
  float* sg_f  = (float*)bump(RD * 4);       // sout, later sop
  float* anew  = (float*)bump(RD * 4);
  float* ab_f  = (float*)bump((size_t)Rc * HIDc * 4);
  float* qf = big4, *kf = big4 + RD, *vf = big4 + 2 * RD, *gf = big4 + 3 * RD;
  float* sw_f = big4;

  const dim3 eb(256);
  auto G1 = [](int n) { return dim3((unsigned)((n + 255) / 256)); };
  const dim3 gemmD(Dc / 128, Rc / 64);       // N=768
  const dim3 gemmH(HIDc / 128, Rc / 64);     // N=1536
  const dim3 gemmS(2 * HIDc / 128, Rc / 64); // N=3072
  const dim3 gblk(256);

  // --- weight prep: fp32 -> bf16, transposed ---
  cvt_transpose_kernel<<<G1(Dc * Dc), eb, 0, stream>>>(ssw1, wt_ssw, Dc, Dc);
  cvt_transpose_kernel<<<G1(Dc * Dc), eb, 0, stream>>>(sbw1, wt_sbw, Dc, Dc);
  cvt_transpose_kernel<<<G1(Dc * Dc), eb, 0, stream>>>(q_w, wt_q, Dc, Dc);
  cvt_transpose_kernel<<<G1(Dc * Dc), eb, 0, stream>>>(k_w, wt_k, Dc, Dc);
  cvt_transpose_kernel<<<G1(Dc * Dc), eb, 0, stream>>>(v_w, wt_v, Dc, Dc);
  cvt_transpose_kernel<<<G1(Dc * Dc), eb, 0, stream>>>(g_w, wt_g, Dc, Dc);
  cvt_transpose_kernel<<<G1(Dc * Dc), eb, 0, stream>>>(o_w, wt_o, Dc, Dc);
  cvt_transpose_kernel<<<G1(Dc * Dc), eb, 0, stream>>>(outproj_w, wt_outp, Dc, Dc);
  cvt_transpose_kernel<<<G1(Dc * Dc), eb, 0, stream>>>(ssw2, wt_ssw2, Dc, Dc);
  cvt_transpose_kernel<<<G1(Dc * Dc), eb, 0, stream>>>(sbw2, wt_sbw2, Dc, Dc);
  cvt_transpose_kernel<<<G1(Dc * Dc), eb, 0, stream>>>(op_w, wt_op, Dc, Dc);
  cvt_transpose_kernel<<<G1(Dc * 2 * HIDc), eb, 0, stream>>>(swish_w, wt_swish, Dc, 2 * HIDc);
  cvt_transpose_kernel<<<G1(Dc * HIDc), eb, 0, stream>>>(a2b_w, wt_a2b, Dc, HIDc);
  cvt_transpose_kernel<<<G1(HIDc * Dc), eb, 0, stream>>>(b2a_w, wt_b2a, HIDc, Dc);
  cvt_bf16_kernel<<<G1((int)RD), eb, 0, stream>>>(s_in, s_bf, (int)RD);

  // --- adaLN 1: b = LN(a)*sigmoid(LN(s,w)@ssw+ssb) + LN(s,w)@sbw ---
  ln_kernel<<<dim3(Rc / 8), eb, 0, stream>>>(a_in, nullptr, an_f, nullptr, Rc, Dc);
  ln_kernel<<<dim3(Rc / 8), eb, 0, stream>>>(s_in, snw1, nullptr, sn_bf, Rc, Dc);
  wmma_gemm_bf16<<<gemmD, gblk, 0, stream>>>(sn_bf, wt_ssw, ssb1, ss_f, Rc, Dc, Dc);
  wmma_gemm_bf16<<<gemmD, gblk, 0, stream>>>(sn_bf, wt_sbw, nullptr, sb_f, Rc, Dc, Dc);
  adaln_combine_kernel<<<G1((int)RD), eb, 0, stream>>>(an_f, ss_f, sb_f, b_bf, (int)RD);

  // --- q/k/v/g projections ---
  wmma_gemm_bf16<<<gemmD, gblk, 0, stream>>>(b_bf, wt_q, q_b, qf, Rc, Dc, Dc);
  wmma_gemm_bf16<<<gemmD, gblk, 0, stream>>>(b_bf, wt_k, nullptr, kf, Rc, Dc, Dc);
  wmma_gemm_bf16<<<gemmD, gblk, 0, stream>>>(b_bf, wt_v, nullptr, vf, Rc, Dc, Dc);
  wmma_gemm_bf16<<<gemmD, gblk, 0, stream>>>(b_bf, wt_g, nullptr, gf, Rc, Dc, Dc);
  repack_qkv_kernel<<<G1(Bc * Hc * Nc * 64), eb, 0, stream>>>(
      qf, kf, vf, q_bf, k_bf, v_t, Bc, Hc, Nc, 0.14433756729740643f);  // 48^-0.5

  // --- attention ---
  attn_wmma<<<dim3(Nc / 16, Hc * Bc), dim3(128), 0, stream>>>(
      q_bf, k_bf, v_t, z_in, o_bf, Bc, Hc, Nc);

  // --- x = (o * sigmoid(b@g_w)) @ o_w ; a_new = a + sigmoid(s@outproj+b)*x ---
  gate_mul_kernel<<<G1((int)RD), eb, 0, stream>>>(o_bf, gf, xin_bf, (int)RD);
  wmma_gemm_bf16<<<gemmD, gblk, 0, stream>>>(xin_bf, wt_o, nullptr, x_f, Rc, Dc, Dc);
  wmma_gemm_bf16<<<gemmD, gblk, 0, stream>>>(s_bf, wt_outp, outproj_b, sg_f, Rc, Dc, Dc);
  residual_gate_kernel<<<G1((int)RD), eb, 0, stream>>>(a_in, sg_f, x_f, anew, (int)RD);

  // --- adaLN 2 ---
  ln_kernel<<<dim3(Rc / 8), eb, 0, stream>>>(anew, nullptr, an_f, nullptr, Rc, Dc);
  ln_kernel<<<dim3(Rc / 8), eb, 0, stream>>>(s_in, snw2, nullptr, sn_bf, Rc, Dc);
  wmma_gemm_bf16<<<gemmD, gblk, 0, stream>>>(sn_bf, wt_ssw2, ssb2, ss_f, Rc, Dc, Dc);
  wmma_gemm_bf16<<<gemmD, gblk, 0, stream>>>(sn_bf, wt_sbw2, nullptr, sb_f, Rc, Dc, Dc);
  adaln_combine_kernel<<<G1((int)RD), eb, 0, stream>>>(an_f, ss_f, sb_f, b_bf, (int)RD);

  // --- SwiGLU FFN ---
  wmma_gemm_bf16<<<gemmS, gblk, 0, stream>>>(b_bf, wt_swish, nullptr, sw_f, Rc, 2 * HIDc, Dc);
  wmma_gemm_bf16<<<gemmH, gblk, 0, stream>>>(b_bf, wt_a2b, nullptr, ab_f, Rc, HIDc, Dc);
  ffn_combine_kernel<<<G1(Rc * HIDc), eb, 0, stream>>>(sw_f, ab_f, hid_bf, Rc, HIDc);
  wmma_gemm_bf16<<<gemmD, gblk, 0, stream>>>(hid_bf, wt_b2a, nullptr, x_f, Rc, Dc, HIDc);
  wmma_gemm_bf16<<<gemmD, gblk, 0, stream>>>(s_bf, wt_op, op_b, sg_f, Rc, Dc, Dc);
  residual_gate_kernel<<<G1((int)RD), eb, 0, stream>>>(anew, sg_f, x_f, out, (int)RD);
}
